// FeedForwardQuantum_22582938042549
// MI455X (gfx1250) — matrix operand order repaired and checked by reading, not compiled, not run on hardware
//
#include <hip/hip_runtime.h>
#include <hip/hip_bf16.h>

typedef __attribute__((ext_vector_type(16))) _Float16 v16h;
typedef __attribute__((ext_vector_type(8)))  float    v8f;

#define NQ   8
#define DIM  256
#define TOKS_PER_BLOCK 16

// ---------------------------------------------------------------------------
// Kernel 0: build the fixed circuit unitary U(qw) (256x256 complex) and pack
// B = [Re(U*diag((-i)^pop)); Im(...)] (512x256) into WMMA B-fragment order.
// Single block, 256 threads; negligible cost (~10 MFLOP).
// ---------------------------------------------------------------------------
__global__ __launch_bounds__(256) void qffn_build_stage(
    const float* __restrict__ qw,   // (2,8)
    float* __restrict__ Ure, float* __restrict__ Uim,   // 256x256 each
    _Float16* __restrict__ Bfrag)   // 32 ctiles * 8 ksteps * 32 lanes * 16 halves
{
  const int tid = threadIdx.x;

  // identity init
  for (int i = tid; i < 65536; i += 256) {
    Ure[i] = ((i >> 8) == (i & 255)) ? 1.0f : 0.0f;
    Uim[i] = 0.0f;
  }
  __syncthreads();

  for (int l = 0; l < 2; ++l) {
    // RX layer (fixed angles qw[l][q]); left-multiply onto U: mixes row pairs
    for (int q = 0; q < NQ; ++q) {
      const float th = qw[l * NQ + q] * 0.5f;
      const float cc = cosf(th), ss = sinf(th);
      const int mask = 0x80 >> q;  // qubit q lives at bit (7-q)
      for (int it = tid; it < 32768; it += 256) {
        const int pr = it >> 8, j = it & 255;
        const int lo = pr & (mask - 1);
        const int i0 = ((pr ^ lo) << 1) | lo;   // row with bit clear
        const int i1 = i0 | mask;
        const float r0 = Ure[i0 * 256 + j], m0 = Uim[i0 * 256 + j];
        const float r1 = Ure[i1 * 256 + j], m1 = Uim[i1 * 256 + j];
        // new0 = c*a0 - i s*a1 ; new1 = c*a1 - i s*a0
        Ure[i0 * 256 + j] = cc * r0 + ss * m1;
        Uim[i0 * 256 + j] = cc * m0 - ss * r1;
        Ure[i1 * 256 + j] = cc * r1 + ss * m0;
        Uim[i1 * 256 + j] = cc * m1 - ss * r0;
      }
      __syncthreads();
    }
    // CNOT ring: left-multiply = swap row i <-> i^tgtmask where ctrl bit set
    for (int q = 0; q < NQ; ++q) {
      const int cm = 0x80 >> q;
      const int tm = 0x80 >> ((q + 1) & 7);
      for (int it = tid; it < 65536; it += 256) {
        const int i = it >> 8, j = it & 255;
        if ((i & cm) && !(i & tm)) {
          const int i2 = i | tm;
          float a = Ure[i * 256 + j]; Ure[i * 256 + j] = Ure[i2 * 256 + j]; Ure[i2 * 256 + j] = a;
          float b = Uim[i * 256 + j]; Uim[i * 256 + j] = Uim[i2 * 256 + j]; Uim[i2 * 256 + j] = b;
        }
      }
      __syncthreads();
    }
  }

  // Pack V = U * diag((-i)^popcount(col)) into f16 WMMA B fragments.
  // Fragment element [ctile][kstep][lane][e]: N=ctile*16+(lane&15),
  // K=kstep*32+(lane>>4)*16+e  (contiguous k order; A fragment uses the same).
  for (int idx = tid; idx < 32 * 8 * 32 * 16; idx += 256) {
    const int e     = idx & 15;
    const int lane  = (idx >> 4) & 31;
    const int kstep = (idx >> 9) & 7;
    const int ctile = idx >> 12;
    const int c = ctile * 16 + (lane & 15);          // 0..511 (re rows then im rows)
    const int k = kstep * 32 + (lane >> 4) * 16 + e; // 0..255
    const int cc = c & 255;
    const float ur = Ure[cc * 256 + k], ui = Uim[cc * 256 + k];
    const int pm = __popc(k) & 3;                    // phase (-i)^pop
    const float pr = (pm == 0) ? 1.f : ((pm == 2) ? -1.f : 0.f);
    const float pi = (pm == 1) ? -1.f : ((pm == 3) ? 1.f : 0.f);
    const float vre = ur * pr - ui * pi;
    const float vim = ur * pi + ui * pr;
    Bfrag[idx] = (_Float16)((c < 256) ? vre : vim);
  }
}

// ---------------------------------------------------------------------------
// Kernel 1: fused per-16-token tile:
//   x -> h -> (cos,sin) -> r (f16) -> psi = R * B^T via v_wmma_f32_16x16x32_f16
//   -> probs -> z -> out = z*W2^T + b2
// ---------------------------------------------------------------------------
__global__ __launch_bounds__(256) void qffn_main(
    const float* __restrict__ x,
    const float* __restrict__ W1, const float* __restrict__ b1,
    const float* __restrict__ W2, const float* __restrict__ b2,
    const _Float16* __restrict__ Bfrag,
    float* __restrict__ out)
{
  __shared__ float    xs[TOKS_PER_BLOCK][256];   // 16 KB
  __shared__ float    cs[TOKS_PER_BLOCK][NQ];
  __shared__ float    sn[TOKS_PER_BLOCK][NQ];
  __shared__ _Float16 rA[TOKS_PER_BLOCK][256];   // 8 KB (WMMA A source)
  __shared__ float    psi[TOKS_PER_BLOCK][512];  // 32 KB (re | im)
  __shared__ float    zz[TOKS_PER_BLOCK][NQ];

  const int tid  = threadIdx.x;
  const long tok0 = (long)blockIdx.x * TOKS_PER_BLOCK;

  // --- 1. stage x tile into LDS (coalesced float4) ---
  {
    const float4* xin = (const float4*)(x + tok0 * 256);
    float4* xls = (float4*)&xs[0][0];
    for (int i = tid; i < TOKS_PER_BLOCK * 64; i += 256) xls[i] = xin[i];
  }
  __syncthreads();

  // --- 2. h = x*W1^T + b1 ; cos/sin of h/2 ---
  if (tid < TOKS_PER_BLOCK * NQ) {
    const int t = tid >> 3, q = tid & 7;
    const float* wrow = W1 + q * 256;
    float acc = b1[q];
    for (int e = 0; e < 256; ++e) acc += xs[t][e] * wrow[e];
    const float ha = acc * 0.5f;
    cs[t][q] = cosf(ha);
    sn[t][q] = sinf(ha);
  }
  __syncthreads();

  // --- 3. r_k = prod_q (bit_q(k) ? sin : cos), f16 ---
  for (int i = tid; i < TOKS_PER_BLOCK * 256; i += 256) {
    const int t = i >> 8, k = i & 255;
    float p = 1.0f;
#pragma unroll
    for (int q = 0; q < NQ; ++q) {
      const int bit = (k >> (7 - q)) & 1;
      p *= bit ? sn[t][q] : cs[t][q];
    }
    rA[t][k] = (_Float16)p;
  }
  __syncthreads();

  // --- 4. psi(16x512) = R(16x256) * B^T via WMMA, K in 8 steps of 32 ---
  const int w = tid >> 5, lane = tid & 31;
  const int mrow = lane & 15, lhalf = lane >> 4;
  v8f acc[4] = {v8f{}, v8f{}, v8f{}, v8f{}};
  for (int ks = 0; ks < 8; ++ks) {
    // A fragment: M=mrow, K=ks*32+(lane>>4)*16+e (matches Bfrag packing)
    v16h a;
#pragma unroll
    for (int e = 0; e < 16; ++e) {
      const int kk = ks * 32 + lhalf * 16 + e;
      a[e] = rA[mrow][kk];
    }
    const v16h* bp = (const v16h*)(Bfrag + ((w * 4 * 8 + ks) * 32 + lane) * 16);
#pragma unroll
    for (int i = 0; i < 4; ++i) {
      const v16h b = bp[i * 256];  // next ctile = +8*32*16 halves = +256 v16h
      acc[i] = __builtin_amdgcn_wmma_f32_16x16x32_f16(
          false, a, false, b, (short)0, acc[i], false, false);
    }
  }
#pragma unroll
  for (int i = 0; i < 4; ++i) {
    const int c = (w * 4 + i) * 16 + mrow;
#pragma unroll
    for (int v = 0; v < 8; ++v) {
      psi[v + 8 * lhalf][c] = acc[i][v];   // D layout: M = v + 8*(lane/16)
    }
  }
  __syncthreads();

  // --- 5. z_q = sum_k sign * (psi_re^2 + psi_im^2) ---
  if (tid < TOKS_PER_BLOCK * NQ) {
    const int t = tid >> 3, q = tid & 7;
    const int sh = 7 - q;
    float s = 0.0f;
    for (int k = 0; k < 256; ++k) {
      const float re = psi[t][k], im = psi[t][k + 256];
      const float p = re * re + im * im;
      s += ((k >> sh) & 1) ? -p : p;
    }
    zz[t][q] = s;
  }
  __syncthreads();

  // --- 6. out = z*W2^T + b2 (coalesced float4 stores) ---
  {
    const int t  = tid >> 4;
    const int e0 = (tid & 15) * 16;
    float zq[NQ];
#pragma unroll
    for (int q = 0; q < NQ; ++q) zq[q] = zz[t][q];
    float* op = out + (tok0 + t) * 256;
    for (int e = e0; e < e0 + 16; e += 4) {
      float4 o;
      float* oo = (float*)&o;
#pragma unroll
      for (int j = 0; j < 4; ++j) {
        const float* wr = W2 + (e + j) * NQ;
        float s = b2[e + j];
#pragma unroll
        for (int q = 0; q < NQ; ++q) s += zq[q] * wr[q];
        oo[j] = s;
      }
      *(float4*)(op + e) = o;
    }
  }
}

extern "C" void kernel_launch(void* const* d_in, const int* in_sizes, int n_in,
                              void* d_out, int out_size, void* d_ws, size_t ws_size,
                              hipStream_t stream) {
  const float* x  = (const float*)d_in[0];   // (32,2048,256)
  const float* W1 = (const float*)d_in[1];   // (8,256)
  const float* b1 = (const float*)d_in[2];   // (8,)
  const float* qw = (const float*)d_in[3];   // (2,8)
  const float* W2 = (const float*)d_in[4];   // (256,8)
  const float* b2 = (const float*)d_in[5];   // (256,)
  float* out = (float*)d_out;

  float* Ure = (float*)d_ws;                 // 256 KB
  float* Uim = Ure + 65536;                  // 256 KB
  _Float16* Bfrag = (_Float16*)(Uim + 65536);// 256 KB

  qffn_build_stage<<<1, 256, 0, stream>>>(qw, Ure, Uim, Bfrag);

  const int nblocks = (32 * 2048) / TOKS_PER_BLOCK;  // 4096
  qffn_main<<<nblocks, 256, 0, stream>>>(x, W1, b1, W2, b2, Bfrag, out);
}
